// RectAxialAttentionBlock_20598663152287
// MI455X (gfx1250) — matrix operand order repaired and checked by reading, not compiled, not run on hardware
//
#include <hip/hip_runtime.h>
#include <hip/hip_bf16.h>

// MI455X (gfx1250) implementation. Bandwidth-bound problem (~15 GFLOP vs
// ~50MB mandatory HBM traffic at 23.3 TB/s): keep intermediates in L2 as
// bf16, do all dense math with v_wmma_f32_16x16x32_bf16 (wave32 layouts per
// CDNA5 ISA 7.12.2), fuse BN/ReLU/softmax epilogues, and move GEMM B-tiles
// with the Tensor Data Mover (TENSORcnt + s_wait_tensorcnt) so tile copies
// run on the DMA engine while waves do WMMA.

typedef __attribute__((ext_vector_type(16))) __bf16 v16bf;
typedef __attribute__((ext_vector_type(8)))  float  v8f;
typedef __attribute__((ext_vector_type(4))) unsigned int tdm_v4u;
typedef __attribute__((ext_vector_type(8))) int          tdm_v8i;
typedef __attribute__((ext_vector_type(4))) int          tdm_v4i;

#define EPSV 1e-5f

#if defined(__HIP_DEVICE_COMPILE__) && __has_builtin(__builtin_amdgcn_tensor_load_to_lds)
#define GEMM_USE_TDM 1
#else
#define GEMM_USE_TDM 0
#endif

__device__ __forceinline__ v8f wmma_bf16(v16bf a, v16bf b, v8f c) {
  return __builtin_amdgcn_wmma_f32_16x16x32_bf16(false, a, false, b, (short)0, c, false, false);
}

// ---------------------------------------------------------------------------
// A-fragment (16x32 bf16, MxK) straight from a row-major global matrix.
// lane: m = lane&15 for both halves; element e -> K = half*16 + e
// => one contiguous 32B read per lane.
__device__ __forceinline__ v16bf load_a_global(const __bf16* __restrict__ A,
                                               int lda, int row0, int k0, int lane) {
  const __bf16* p = A + (row0 + (lane & 15)) * lda + k0 + (lane >> 4) * 16;
  v16bf a;
#pragma unroll
  for (int e = 0; e < 8; ++e) { a[e] = p[e]; a[8 + e] = p[8 + e]; }
  return a;
}

// B-fragment from LDS tile stored transposed [n][KT+4] (gather GEMM).
__device__ __forceinline__ v16bf load_b_lds_t(const __bf16* __restrict__ base, int lane) {
  const __bf16* p = base + (lane & 15) * 36 + (lane >> 4) * 16;
  v16bf b;
#pragma unroll
  for (int e = 0; e < 16; ++e) b[e] = p[e];
  return b;
}

// B-fragment from LDS tile stored k-major [KT][128] (TDM deposit order).
// lane: n = lane&15 (+tile offset in base); element e -> K = half*16 + e.
__device__ __forceinline__ v16bf load_b_lds_k(const __bf16* __restrict__ base, int lane) {
  const __bf16* p = base + (lane & 15) + (lane >> 4) * 16 * 128;
  v16bf b;
#pragma unroll
  for (int e = 0; e < 16; ++e) b[e] = p[e * 128];
  return b;
}

// ---------------------------------------------------------------------------
// TDM: DMA a [rows=32][cols=128] bf16 tile (row stride = 24576 elements) from
// global into LDS (packed k-major). D# per CDNA5 ISA 8.3/8.4:
//  g0: count=1 | lds_addr | global_addr[56:0] | type=2
//  g1: data_size=1(2B) | tensor_dim0=S | tensor_dim1=rows | tile_dim0=128 |
//      tile_dim1=32 | tensor_dim0_stride=S
__device__ __forceinline__ void tdm_load_tile(unsigned lds_addr, const __bf16* gptr) {
#if GEMM_USE_TDM
  constexpr unsigned S = 24576, ROWS = 32, COLS = 128;
  unsigned long long ga = (unsigned long long)(uintptr_t)gptr;
  tdm_v4u g0;
  g0[0] = 1u;                                            // count=1 (valid)
  g0[1] = lds_addr;                                      // LDS byte address
  g0[2] = (unsigned)ga;                                  // global_addr[31:0]
  g0[3] = (unsigned)((ga >> 32) & 0x01FFFFFFu) | (2u << 30);   // [56:32], type=2
  tdm_v8i g1;
  g1[0] = (int)(1u << 16);                               // data_size=1 -> 2 bytes
  g1[1] = (int)((S & 0xFFFFu) << 16);                    // tensor_dim0[15:0]
  g1[2] = (int)(((S >> 16) & 0xFFFFu) | ((ROWS & 0xFFFFu) << 16)); // d0 hi | d1 lo
  g1[3] = (int)((COLS & 0xFFFFu) << 16);                 // tile_dim0
  g1[4] = (int)(ROWS & 0xFFFFu);                         // tile_dim1 (tile_dim2=0)
  g1[5] = (int)S;                                        // tensor_dim0_stride[31:0]
  g1[6] = 0;                                             // stride hi | dim1_stride lo
  g1[7] = 0;
  tdm_v4i gz = {};
#if __clang_major__ >= 23
  tdm_v8i gz8 = {};
  __builtin_amdgcn_tensor_load_to_lds(g0, g1, gz, gz, gz8, 0);
#else
  __builtin_amdgcn_tensor_load_to_lds(g0, g1, gz, gz, 0);
#endif
#else
  (void)lds_addr; (void)gptr;
#endif
}

// ---------------------------------------------------------------------------
// WMMA GEMM with contiguous bf16 B [K][24576]: TDM double-buffered staging.
// Block tile 64x128, 8 waves of 32x32.
template <typename Epi>
__global__ __launch_bounds__(256) void gemm_wmma_tdm(const __bf16* __restrict__ A,
                                                     int K,
                                                     const __bf16* __restrict__ Bsrc,
                                                     Epi epi) {
  constexpr int BM = 64, BN = 128, KT = 32, S = 24576;
  __shared__ __bf16 Bt[2][KT][BN];

  const int tid  = threadIdx.x;
  const int lane = tid & 31;
  const int wid  = tid >> 5;
  const int wm   = wid >> 2;
  const int wn   = wid & 3;
  const int nBlock = blockIdx.x * BN;
  const int mBase  = blockIdx.y * BM + wm * 32;

  v8f acc[2][2] = {};

#if GEMM_USE_TDM
  const unsigned lds0 = (unsigned)(unsigned long long)(uintptr_t)&Bt[0][0][0];
  const unsigned lds1 = (unsigned)(unsigned long long)(uintptr_t)&Bt[1][0][0];
  if (tid < 32) {                       // wave 0 drives the DMA engine
    tdm_load_tile(lds0, Bsrc + nBlock);
    __builtin_amdgcn_s_wait_tensorcnt((short)0);
  }
#else
  // Fallback: vectorized b128 global -> b128 LDS copies (8 bf16 per op).
  for (int t = tid; t < KT * BN / 8; t += 256) {
    int k = t >> 4, c = t & 15;
    ((int4*)&Bt[0][k][0])[c] =
        ((const int4*)(Bsrc + (size_t)k * S + nBlock))[c];
  }
#endif
  __syncthreads();

  int buf = 0;
  for (int k0 = 0; k0 < K; k0 += KT) {
    if (k0 + KT < K) {
#if GEMM_USE_TDM
      if (tid < 32)
        tdm_load_tile(buf ? lds0 : lds1, Bsrc + (size_t)(k0 + KT) * S + nBlock);
#else
      for (int t = tid; t < KT * BN / 8; t += 256) {
        int k = t >> 4, c = t & 15;
        ((int4*)&Bt[buf ^ 1][k][0])[c] =
            ((const int4*)(Bsrc + (size_t)(k0 + KT + k) * S + nBlock))[c];
      }
#endif
    }
    v16bf a0 = load_a_global(A, K, mBase,      k0, lane);
    v16bf a1 = load_a_global(A, K, mBase + 16, k0, lane);
    v16bf b0 = load_b_lds_k(&Bt[buf][0][wn * 32],      lane);
    v16bf b1 = load_b_lds_k(&Bt[buf][0][wn * 32 + 16], lane);
    acc[0][0] = wmma_bf16(a0, b0, acc[0][0]);
    acc[0][1] = wmma_bf16(a0, b1, acc[0][1]);
    acc[1][0] = wmma_bf16(a1, b0, acc[1][0]);
    acc[1][1] = wmma_bf16(a1, b1, acc[1][1]);
#if GEMM_USE_TDM
    if (tid < 32) __builtin_amdgcn_s_wait_tensorcnt((short)0);
#endif
    __syncthreads();
    buf ^= 1;
  }

  const int half = lane >> 4, nq = lane & 15;
#pragma unroll
  for (int i = 0; i < 2; ++i)
#pragma unroll
    for (int j = 0; j < 2; ++j) {
      int m0 = mBase + i * 16 + half * 8;
      int n0 = nBlock + wn * 32 + j * 16 + nq;
#pragma unroll
      for (int r = 0; r < 8; ++r) epi(m0 + r, n0, acc[i][j][r]);
    }
}

// ---------------------------------------------------------------------------
// Generic gather-B WMMA GEMM (used for conv_in whose B is a strided f32
// gather with on-the-fly bf16 conversion). LDS tile transposed [n][KT+4].
template <typename BLoad, typename Epi>
__global__ __launch_bounds__(256) void gemm_wmma(const __bf16* __restrict__ A,
                                                 int K, BLoad bload, Epi epi) {
  constexpr int BN = 128, KT = 32;
  __shared__ __bf16 Bt[2][BN][KT + 4];

  const int tid  = threadIdx.x;
  const int lane = tid & 31;
  const int wid  = tid >> 5;
  const int wm   = wid >> 2;
  const int wn   = wid & 3;
  const int nBlock = blockIdx.x * BN;
  const int mBase  = blockIdx.y * 64 + wm * 32;

  v8f acc[2][2] = {};

  auto loadB = [&](int buf, int k0) {
#pragma unroll 4
    for (int t = tid; t < BN * KT; t += 256) {
      int k = t / BN, nn = t % BN;            // global reads coalesced along n
      Bt[buf][nn][k] = bload(k0 + k, nBlock + nn);
    }
  };

  loadB(0, 0);
  __syncthreads();

  int buf = 0;
  for (int k0 = 0; k0 < K; k0 += KT) {
    if (k0 + KT < K) loadB(buf ^ 1, k0 + KT);
    v16bf a0 = load_a_global(A, K, mBase,      k0, lane);
    v16bf a1 = load_a_global(A, K, mBase + 16, k0, lane);
    v16bf b0 = load_b_lds_t(&Bt[buf][wn * 32][0], lane);
    v16bf b1 = load_b_lds_t(&Bt[buf][wn * 32 + 16][0], lane);
    acc[0][0] = wmma_bf16(a0, b0, acc[0][0]);
    acc[0][1] = wmma_bf16(a0, b1, acc[0][1]);
    acc[1][0] = wmma_bf16(a1, b0, acc[1][0]);
    acc[1][1] = wmma_bf16(a1, b1, acc[1][1]);
    __syncthreads();
    buf ^= 1;
  }

  const int half = lane >> 4, nq = lane & 15;
#pragma unroll
  for (int i = 0; i < 2; ++i)
#pragma unroll
    for (int j = 0; j < 2; ++j) {
      int m0 = mBase + i * 16 + half * 8;
      int n0 = nBlock + wn * 32 + j * 16 + nq;
#pragma unroll
      for (int r = 0; r < 8; ++r) epi(m0 + r, n0, acc[i][j][r]);
    }
}

// ---------------------------------------------------------------------------
// Functors (B = 2, C_IN = 256, H = 96, W = 128, D_IN = 128, S = 24576)
struct BLoadConvIn {            // B(k=c, n=sH), sH = (b*W+w)*H + h
  const float* x;
  __device__ __bf16 operator()(int c, int s) const {
    int n = s / 96, h = s % 96;
    int b = n >> 7, w = n & 127;
    return (__bf16)x[((b * 256 + c) * 96 + h) * 128 + w];
  }
};
struct EpiConvIn {              // relu(bn_in(v)) -> bf16 [c=128][sH]
  __bf16* out; const float* bn;
  __device__ void operator()(int m, int s, float v) const {
    float sc = bn[m] * rsqrtf(bn[384 + m] + EPSV);
    float y  = (v - bn[256 + m]) * sc + bn[128 + m];
    out[m * 24576 + s] = (__bf16)(y > 0.f ? y : 0.f);
  }
};
struct EpiStoreF32 {            // raw qkv projection -> f32 [o=256][S]
  float* out;
  __device__ void operator()(int m, int s, float v) const { out[m * 24576 + s] = v; }
};
struct EpiConvOut {             // relu(bn_out(v) + x_in), s = b*HW + hw
  float* out; const float* bn; const float* x;
  __device__ void operator()(int m, int s, float v) const {
    float sc = bn[m] * rsqrtf(bn[768 + m] + EPSV);
    float y  = (v - bn[512 + m]) * sc + bn[256 + m];
    int b = s / 12288, hw = s % 12288;
    int idx = (b * 256 + m) * 12288 + hw;
    y += x[idx];
    out[idx] = y > 0.f ? y : 0.f;
  }
};

// ---------------------------------------------------------------------------
// Fused axial attention: one workgroup (128 threads / 4 waves) per (seq, head).
// HEADS=8, DHKQ=8, DHV=16. qkv layout: [o=256][S], o = qkv_ch*8 + head.
template <int L, bool HPASS, bool RELU>
__global__ __launch_bounds__(128) void axial_attn(
    const float* __restrict__ qkv, const float* __restrict__ rel,
    const float* __restrict__ attnbn, const float* __restrict__ outbn,
    __bf16* __restrict__ yout) {
  constexpr int R  = 2 * L - 1;
  constexpr int VP = L + 4;     // pad to break LDS bank alignment
  constexpr int SP = L + 8;
  constexpr int LT = L / 16;
  constexpr int S  = 24576;

  extern __shared__ char smem_raw[];
  float* qf   = (float*)smem_raw;        // [8][VP]
  float* kf   = qf   + 8 * VP;           // [8][VP]
  float* vf   = kf   + 8 * VP;           // [16][VP]
  float* rq   = vf   + 16 * VP;          // [8][R]
  float* rk   = rq   + 8 * R;            // [8][R]
  float* rv   = rk   + 8 * R;            // [16][R]
  float* sc   = rv   + 16 * R;           // [L][SP]  scores -> probs
  float* kvf  = sc   + L * SP;           // [16][VP]
  float* outT = kvf  + 16 * VP;          // [L][17]

  const int tid  = threadIdx.x;
  const int lane = tid & 31;
  const int wid  = tid >> 5;
  const int n    = blockIdx.x >> 3;
  const int head = blockIdx.x & 7;

  // ---- stage q,k,v and relative embeddings into LDS --------------------
  for (int t = tid; t < 8 * L; t += 128) {
    int i = t / L, l = t % L;
    qf[i * VP + l] = qkv[(i * 8 + head) * S + n * L + l];
    kf[i * VP + l] = qkv[((8 + i) * 8 + head) * S + n * L + l];
  }
  for (int t = tid; t < 16 * L; t += 128) {
    int i = t / L, l = t % L;
    vf[i * VP + l] = qkv[((16 + i) * 8 + head) * S + n * L + l];
  }
  for (int t = tid; t < 8 * R; t += 128) {
    int i = t / R, r = t % R;
    rq[i * R + r] = rel[i * R + r];
    rk[i * R + r] = rel[(8 + i) * R + r];
  }
  for (int t = tid; t < 16 * R; t += 128) {
    int i = t / R, r = t % R;
    rv[i * R + r] = rel[(16 + i) * R + r];
  }
  __syncthreads();

  // ---- dots = q^T k via WMMA (K=8 zero-padded inside one K=32 op) ------
  {
    const int half = lane >> 4, q = lane & 15;
    for (int t = wid; t < LT * LT; t += 4) {
      int lt = t / LT, jt = t % LT;
      v16bf a = {}, b = {};
      if (half == 0) {            // half==1 lanes map K>=8 -> stay zero
        int l = lt * 16 + q, j = jt * 16 + q;
#pragma unroll
        for (int e = 0; e < 8; ++e) {
          a[e] = (__bf16)qf[e * VP + l];   // A[m=l][K=i] = q[i,l]
          b[e] = (__bf16)kf[e * VP + j];   // B[K=i][n=j] = k[i,j]
        }
      }
      v8f c = {};
      c = wmma_bf16(a, b, c);
      int jg = jt * 16 + q;
#pragma unroll
      for (int r = 0; r < 8; ++r) sc[(lt * 16 + r + half * 8) * SP + jg] = c[r];
    }
  }
  __syncthreads();

  // ---- s = bn(qr) + bn(kr) + bn(dots)  (attnbn is [4][24]) -------------
  {
    const int ch = head * 3;
    float sQ = attnbn[ch]     * rsqrtf(attnbn[72 + ch]     + EPSV);
    float cQ = attnbn[24 + ch]     - attnbn[48 + ch]     * sQ;
    float sK = attnbn[ch + 1] * rsqrtf(attnbn[72 + ch + 1] + EPSV);
    float cK = attnbn[24 + ch + 1] - attnbn[48 + ch + 1] * sK;
    float sD = attnbn[ch + 2] * rsqrtf(attnbn[72 + ch + 2] + EPSV);
    float cD = attnbn[24 + ch + 2] - attnbn[48 + ch + 2] * sD;
    for (int t = tid; t < L * L; t += 128) {
      int l = t / L, j = t % L;
      int r = j - l + (L - 1);
      float qr = 0.f, kr = 0.f;
#pragma unroll
      for (int i = 0; i < 8; ++i) {
        qr += qf[i * VP + l] * rq[i * R + r];
        kr += kf[i * VP + l] * rk[i * R + r];
      }
      sc[l * SP + j] = sQ * qr + cQ + sK * kr + cK + sD * sc[l * SP + j] + cD;
    }
  }
  __syncthreads();

  // ---- row softmax ------------------------------------------------------
  for (int l = tid; l < L; l += 128) {
    float mx = -3.4e38f;
    for (int j = 0; j < L; ++j) mx = fmaxf(mx, sc[l * SP + j]);
    float sum = 0.f;
    for (int j = 0; j < L; ++j) {
      float e = __expf(sc[l * SP + j] - mx);
      sc[l * SP + j] = e;
      sum += e;
    }
    float inv = 1.f / sum;
    for (int j = 0; j < L; ++j) sc[l * SP + j] *= inv;
  }
  __syncthreads();

  // ---- out^T[l,i] = sum_j p[l,j] v[i,j]  via WMMA (K=L) ----------------
  {
    const int half = lane >> 4, q = lane & 15;
    for (int lt = wid; lt < LT; lt += 4) {
      v8f c = {};
      for (int k0 = 0; k0 < L; k0 += 32) {
        v16bf a, b;
        int l = lt * 16 + q;
#pragma unroll
        for (int e = 0; e < 16; ++e) {
          int kk = k0 + half * 16 + e;
          a[e] = (__bf16)sc[l * SP + kk];        // A[m=l][K=j] = p[l,j]
        }
#pragma unroll
        for (int e = 0; e < 16; ++e)
          b[e] = (__bf16)vf[q * VP + k0 + half * 16 + e];  // B[K=j][n=i]=v[i,j]
        c = wmma_bf16(a, b, c);
      }
#pragma unroll
      for (int r = 0; r < 8; ++r) outT[(lt * 16 + r + half * 8) * 17 + q] = c[r];
    }
  }

  // ---- kv[i,l] = sum_j p[l,j] r_v[i, j-l+L-1]  (VALU) ------------------
  for (int t = tid; t < 16 * L; t += 128) {
    int i = t / L, l = t % L;
    const float* prow = &sc[l * SP];
    const float* rrow = &rv[i * R + (L - 1 - l)];
    float s = 0.f;
    for (int j = 0; j < L; ++j) s += prow[j] * rrow[j];
    kvf[i * VP + l] = s;
  }
  __syncthreads();

  // ---- y[d] = bn(kv)[d] + bn(out)[128+d], optional relu, scatter -------
  for (int t = tid; t < 16 * L; t += 128) {
    int i = t / L, l = t % L;
    int d = head * 16 + i, d2 = 128 + d;
    float s1 = outbn[d]  * rsqrtf(outbn[768 + d]  + EPSV);
    float c1 = outbn[256 + d]  - outbn[512 + d]  * s1;
    float s2 = outbn[d2] * rsqrtf(outbn[768 + d2] + EPSV);
    float c2 = outbn[256 + d2] - outbn[512 + d2] * s2;
    float y = s1 * kvf[i * VP + l] + c1 + s2 * outT[l * 17 + i] + c2;
    if (RELU) y = y > 0.f ? y : 0.f;
    int idx;
    if (HPASS) {                 // n = b*128 + w, l = h -> [(b*96+h)*128 + w]
      int b = n >> 7, w = n & 127;
      idx = d * S + (b * 96 + l) * 128 + w;
    } else {                     // n = b*96 + h, l = w
      idx = d * S + n * 128 + l;
    }
    yout[idx] = (__bf16)y;
  }
}

// ---------------------------------------------------------------------------
__global__ __launch_bounds__(256) void cvt_bf16_kernel(const float* __restrict__ in,
                                                       __bf16* __restrict__ out, int n) {
  int i = blockIdx.x * 256 + threadIdx.x;
  if (i < n) out[i] = (__bf16)in[i];
}

static size_t attn_smem_bytes(int L) {
  int VP = L + 4, R = 2 * L - 1, SP = L + 8;
  return (size_t)(48 * VP + 32 * R + L * SP + 17 * L) * sizeof(float);
}

extern "C" void kernel_launch(void* const* d_in, const int* in_sizes, int n_in,
                              void* d_out, int out_size, void* d_ws, size_t ws_size,
                              hipStream_t stream) {
  (void)in_sizes; (void)n_in; (void)out_size; (void)ws_size;

  const float* x_in     = (const float*)d_in[0];
  const float* w_in     = (const float*)d_in[1];
  const float* bn_in    = (const float*)d_in[2];
  const float* w_out    = (const float*)d_in[3];
  const float* bn_out   = (const float*)d_in[4];
  const float* wqkv_h   = (const float*)d_in[5];
  const float* rel_h    = (const float*)d_in[6];
  const float* attnbn_h = (const float*)d_in[7];
  const float* outbn_h  = (const float*)d_in[8];
  const float* wqkv_w   = (const float*)d_in[9];
  const float* rel_w    = (const float*)d_in[10];
  const float* attnbn_w = (const float*)d_in[11];
  const float* outbn_w  = (const float*)d_in[12];

  char* ws = (char*)d_ws;
  __bf16* w_in_bf   = (__bf16*)(ws + 0);          // 128*256
  __bf16* wqkv_h_bf = (__bf16*)(ws + 65536);      // 256*128
  __bf16* wqkv_w_bf = (__bf16*)(ws + 131072);     // 256*128
  __bf16* w_out_bf  = (__bf16*)(ws + 196608);     // 256*128
  __bf16* act1      = (__bf16*)(ws + 262144);     // [128][24576] bf16
  __bf16* actW      = (__bf16*)(ws + 6553600);    // [128][24576] bf16
  __bf16* ywbuf     = (__bf16*)(ws + 12845056);   // [128][24576] bf16
  float*  qkvbuf    = (float*) (ws + 19136512);   // [256][24576] f32

  // weights -> bf16
  cvt_bf16_kernel<<<128, 256, 0, stream>>>(w_in,   w_in_bf,   128 * 256);
  cvt_bf16_kernel<<<128, 256, 0, stream>>>(wqkv_h, wqkv_h_bf, 256 * 128);
  cvt_bf16_kernel<<<128, 256, 0, stream>>>(wqkv_w, wqkv_w_bf, 256 * 128);
  cvt_bf16_kernel<<<128, 256, 0, stream>>>(w_out,  w_out_bf,  256 * 128);

  // conv_in + bn + relu  -> act1 [c][sH]   (gather-B path)
  gemm_wmma<<<dim3(192, 2), 256, 0, stream>>>(w_in_bf, 256,
      BLoadConvIn{x_in}, EpiConvIn{act1, bn_in});
  // qkv projection (H pass) -> qkvbuf f32  (TDM-staged B)
  gemm_wmma_tdm<<<dim3(192, 4), 256, 0, stream>>>(wqkv_h_bf, 128,
      act1, EpiStoreF32{qkvbuf});
  // axial attention along H (L=96, no relu), scatter to W-pass layout
  axial_attn<96, true, false><<<256 * 8, 128, attn_smem_bytes(96), stream>>>(
      qkvbuf, rel_h, attnbn_h, outbn_h, actW);
  // qkv projection (W pass)
  gemm_wmma_tdm<<<dim3(192, 4), 256, 0, stream>>>(wqkv_w_bf, 128,
      actW, EpiStoreF32{qkvbuf});
  // axial attention along W (L=128, relu)
  axial_attn<128, false, true><<<192 * 8, 128, attn_smem_bytes(128), stream>>>(
      qkvbuf, rel_w, attnbn_w, outbn_w, ywbuf);
  // conv_out + bn + residual + relu -> d_out
  gemm_wmma_tdm<<<dim3(192, 4), 256, 0, stream>>>(w_out_bf, 128,
      ywbuf, EpiConvOut{(float*)d_out, bn_out, x_in});
}
